// RenderModel_50603304681862
// MI455X (gfx1250) — matrix operand order repaired and checked by reading, not compiled, not run on hardware
//
#include <hip/hip_runtime.h>
#include <cstdint>
#include <cstddef>

// ---------------------------------------------------------------------------
// CDNA5 (gfx1250, wave32) implementation of the NeRF render model.
// Heavy GEMMs (2D convs via implicit im2col, fused per-point MLP) run on
// v_wmma_f32_16x16x32_bf16. Small convs / BN / elementwise are VALU kernels.
// ---------------------------------------------------------------------------

typedef __attribute__((ext_vector_type(16))) __bf16       bf16x16;
typedef __attribute__((ext_vector_type(8)))  float        v8f;
typedef __attribute__((ext_vector_type(8)))  unsigned int u32x8;

// fp32 -> bf16 bits, round-to-nearest-even
__device__ __forceinline__ unsigned short f2bf(float f) {
  unsigned int u = __builtin_bit_cast(unsigned int, f);
  unsigned int r = u + 0x7fffu + ((u >> 16) & 1u);
  return (unsigned short)(r >> 16);
}
__device__ __forceinline__ float bf2f(unsigned short u) {
  unsigned int x = ((unsigned int)u) << 16;
  return __builtin_bit_cast(float, x);
}

// 16-bit A-matrix 16x32 lane/element -> K mapping (matches B packing):
// lanes 0-15 hold K=0..15, lanes 16-31 hold K=16..31 (element e -> K).
// Per lane this yields one contiguous 16-element run.
__device__ __forceinline__ int amap_k(int lane, int e) {
  return (lane < 16) ? e : 16 + e;
}

// B-matrix 32x16: lanes 0-15 hold K=0..15, lanes 16-31 hold K=16..31 (element e -> K)
__device__ __forceinline__ bf16x16 load_frag16(const unsigned short* p) {
  u32x8 r = *(const u32x8*)p;
  return __builtin_bit_cast(bf16x16, r);
}

__device__ __forceinline__ v8f wmma_bf16(bf16x16 a, bf16x16 b, v8f c) {
  return __builtin_amdgcn_wmma_f32_16x16x32_bf16(false, a, false, b, (short)0, c,
                                                 false, false);
}

// ---------------------------------------------------------------------------
// Weight packer (linear K): W[N,K] fp32 -> per-fragment bf16 layout
// out[((nt*KBpad + kb)*32 + lane)*16 + e] = W[nt*16 + lane%16, kb*32 + bmap(lane,e)]
// ---------------------------------------------------------------------------
__global__ void pack_w(const float* __restrict__ W, unsigned short* __restrict__ out,
                       int K, int KBpad, int total) {
  for (int i = blockIdx.x * blockDim.x + threadIdx.x; i < total;
       i += gridDim.x * blockDim.x) {
    int e    = i & 15;
    int lane = (i >> 4) & 31;
    int kb   = (i >> 9) % KBpad;
    int nt   = i / (KBpad << 9);
    int n = nt * 16 + (lane & 15);
    int k = (kb << 5) + ((lane < 16) ? e : 16 + e);
    float v = (k < K) ? W[(size_t)n * K + k] : 0.f;
    out[i] = f2bf(v);
  }
}

// Conv weight packer, kpos-major K ordering: k' = kpos*256 + ic (IC fixed 256).
// W layout [N][256][9]; value = W[n][ic][kpos].
__global__ void pack_conv(const float* __restrict__ W, unsigned short* __restrict__ out,
                          int KBp, int total) {
  for (int i = blockIdx.x * blockDim.x + threadIdx.x; i < total;
       i += gridDim.x * blockDim.x) {
    int e    = i & 15;
    int lane = (i >> 4) & 31;
    int kb   = (i >> 9) % KBp;
    int nt   = i / (KBp << 9);
    int n = nt * 16 + (lane & 15);
    int k = (kb << 5) + ((lane < 16) ? e : 16 + e);
    int kpos = k >> 8;
    int ic   = k & 255;
    float v = W[((size_t)n * 256 + ic) * 9 + kpos];
    out[i] = f2bf(v);
  }
}

// ---------------------------------------------------------------------------
// Implicit-GEMM conv2d 3x3 SAME via WMMA, IC = 256 fixed (K = 2304, kpos-major).
// Optional nearest 2x upsample of the input (upshift=1) folded into the gather.
// M-tile = 64 output pixels in one row (Wout % 64 == 0): each wave keeps 4
// accumulators and reuses one B fragment across 4 WMMAs. Double-buffered A
// staging in LDS -> one barrier per k-step. TPB = (OC/16)*32 (compile-time).
// ---------------------------------------------------------------------------
template <int TPB>
__global__ void conv2d_wmma(const float* __restrict__ in,
                            const unsigned short* __restrict__ wp,
                            const float* __restrict__ bias, float* __restrict__ out,
                            int Hin, int Win, int OC, int Hout, int Wout,
                            int upshift) {
  const int tid  = threadIdx.x;
  const int lane = tid & 31;
  const int wv   = tid >> 5;
  const int KB   = 72;  // 2304/32
  const int m0   = blockIdx.x * 64;
  const int x0   = m0 % Wout;
  int t          = m0 / Wout;
  const int y    = t % Hout;
  const int b    = t / Hout;
  const int HW   = Hin * Win;

  __shared__ unsigned short Alds[2][64 * 32];  // double-buffered [64 m][32 k]

  v8f c[4];
  {
    float bv = bias[wv * 16 + (lane & 15)];
#pragma unroll
    for (int s = 0; s < 4; ++s)
#pragma unroll
      for (int i = 0; i < 8; ++i) c[s][i] = bv;
  }
  const unsigned short* wbase = wp + ((size_t)wv * KB) * 512 + (size_t)lane * 16;
  const float* inb = in + (size_t)b * 256 * HW;

  auto stage = [&](int kb, int buf) {
    int kpos = kb >> 3;           // scalar: which of the 9 taps
    int icb  = (kb & 7) << 5;     // scalar: ic block base
    int ky   = (kpos >= 6) ? 2 : (kpos >= 3 ? 1 : 0);
    int kx   = kpos - ky * 3;
    int yo   = y + ky - 1;
    bool yok = (unsigned)yo < (unsigned)Hout;
    int yi   = (yo >> upshift) * Win;
#pragma unroll
    for (int j = 0; j < 2048 / TPB; ++j) {
      int idx = tid + j * TPB;
      int mi  = idx >> 5;
      int ic  = icb + (idx & 31);
      int xo  = x0 + mi + kx - 1;
      float v = 0.f;
      if (yok && (unsigned)xo < (unsigned)Wout)
        v = inb[ic * HW + yi + (xo >> upshift)];
      Alds[buf][idx] = f2bf(v);
    }
  };

  stage(0, 0);
  __syncthreads();
  for (int kb = 0; kb < KB; ++kb) {
    int buf = kb & 1;
    if (kb + 1 < KB) stage(kb + 1, buf ^ 1);  // overlaps with WMMA below
    bf16x16 bm = load_frag16(wbase + (size_t)kb * 512);
    bf16x16 a[4];
#pragma unroll
    for (int s = 0; s < 4; ++s) {             // preload all A frags (DS pipelines)
      const int row = ((s * 16) + (lane & 15)) << 5;
#pragma unroll
      for (int e = 0; e < 16; ++e)
        a[s][e] = __builtin_bit_cast(__bf16, Alds[buf][row + amap_k(lane, e)]);
    }
#pragma unroll
    for (int s = 0; s < 4; ++s) c[s] = wmma_bf16(a[s], bm, c[s]);
    __syncthreads();
  }
  // C layout: lane holds n = lane%16, m = v + (lane<16 ? 0 : 8)
  const int n  = wv * 16 + (lane & 15);
  const int mb = (lane < 16) ? 0 : 8;
  float* op = out + (((size_t)b * OC + n) * Hout + y) * Wout + x0 + mb;
#pragma unroll
  for (int s = 0; s < 4; ++s)
#pragma unroll
    for (int v = 0; v < 8; ++v) op[s * 16 + v] = c[s][v];
}

// ---------------------------------------------------------------------------
// Fused per-point NeRF MLP + volume render. One block = 16 points = 2 pixels
// x 8 floors; 8 waves, activations LDS-resident in bf16.
// ---------------------------------------------------------------------------
__device__ __forceinline__ v8f mlp_gemm(const unsigned short* actIn, int stride,
                                        const unsigned short* wp, int KB,
                                        const float* bias, int nt, int lane) {
  v8f c;
  float bv = bias[nt * 16 + (lane & 15)];
#pragma unroll
  for (int i = 0; i < 8; ++i) c[i] = bv;
  const unsigned short* wb = wp + ((size_t)nt * KB) * 512 + (size_t)lane * 16;
  const int row = (lane & 15) * stride;
  for (int kb = 0; kb < KB; ++kb) {
    bf16x16 a;
#pragma unroll
    for (int e = 0; e < 16; ++e)
      a[e] = __builtin_bit_cast(__bf16, actIn[row + (kb << 5) + amap_k(lane, e)]);
    bf16x16 bm = load_frag16(wb + (size_t)kb * 512);
    c = wmma_bf16(a, bm, c);
  }
  return c;
}

__device__ __forceinline__ void frag_store_relu(unsigned short* outb, int stride,
                                                int coloff, v8f c, int nt, int lane,
                                                bool relu) {
  int n  = coloff + nt * 16 + (lane & 15);
  int mb = (lane < 16) ? 0 : 8;
#pragma unroll
  for (int v = 0; v < 8; ++v) {
    float val = c[v];
    if (relu) val = fmaxf(val, 0.f);
    outb[(mb + v) * stride + n] = f2bf(val);
  }
}

__global__ void mlp_render(const float* __restrict__ sig, const float* __restrict__ rf,
                           const unsigned short* P1, const float* b1,
                           const unsigned short* P2, const float* b2,
                           const float* w31, const float* b31,
                           const unsigned short* P32, const float* b32,
                           const unsigned short* P33, const float* b33,
                           const unsigned short* P4, const float* b4,
                           const unsigned short* P5, const float* b5,
                           const unsigned short* P6, const float* b6,
                           float* __restrict__ render) {
  __shared__ unsigned short sA[16 * 160];
  __shared__ unsigned short sB[16 * 128];
  __shared__ float s_sig[16];
  __shared__ float s_w[16];
  const int tid  = threadIdx.x;
  const int lane = tid & 31;
  const int wv   = tid >> 5;
  const int tile = blockIdx.x;

  // gather sigma_in [16 pts][16 ch, padded to 32] -> sB (bf16)
  for (int idx = tid; idx < 512; idx += 256) {
    int mi = idx >> 5, ch = idx & 31;
    float v = 0.f;
    if (ch < 16) {
      int m = tile * 16 + mi;
      int f = m & 7; int pl = m >> 3;
      int x = pl & 63, yy = (pl >> 6) & 63, b = pl >> 12;
      v = sig[((((size_t)b * 16 + ch) * 8 + f) * 64 + yy) * 64 + x];
    }
    sB[mi * 32 + ch] = f2bf(v);
  }
  __syncthreads();
  { v8f c = mlp_gemm(sB, 32, P1, 1, b1, wv, lane);      // L1: 16->128
    frag_store_relu(sA, 128, 0, c, wv, lane, true); }
  __syncthreads();
  { v8f c = mlp_gemm(sA, 128, P2, 4, b2, wv, lane);     // L2: 128->128
    frag_store_relu(sB, 128, 0, c, wv, lane, true); }
  __syncthreads();
  // sigma head (l31, N=1) + volume-render weights
  if (tid < 16) {
    float acc = b31[0];
    for (int k = 0; k < 128; ++k) acc += bf2f(sB[tid * 128 + k]) * w31[k];
    s_sig[tid] = fmaxf(acc, 0.f);
  }
  __syncthreads();
  if (tid < 16) {
    int p = tid >> 3, f = tid & 7;
    float T = 0.f;
    for (int g = 0; g <= f; ++g) T -= s_sig[p * 8 + g];
    s_w[tid] = __expf(T) * (1.f - __expf(-s_sig[tid]));
  }
  if (wv < 2) {                                          // L32: 128->32
    v8f c = mlp_gemm(sB, 128, P32, 4, b32, wv, lane);
    frag_store_relu(sA, 160, 0, c, wv, lane, true);
  }
  __syncthreads();
  // gather rgb_in -> sB
  for (int idx = tid; idx < 512; idx += 256) {
    int mi = idx >> 5, ch = idx & 31;
    float v = 0.f;
    if (ch < 16) {
      int m = tile * 16 + mi;
      int f = m & 7; int pl = m >> 3;
      int x = pl & 63, yy = (pl >> 6) & 63, b = pl >> 12;
      v = rf[((((size_t)b * 16 + ch) * 8 + f) * 64 + yy) * 64 + x];
    }
    sB[mi * 32 + ch] = f2bf(v);
  }
  __syncthreads();
  { v8f c = mlp_gemm(sB, 32, P33, 1, b33, wv, lane);     // L33: 16->128
    frag_store_relu(sA, 160, 32, c, wv, lane, true); }
  __syncthreads();
  { v8f c = mlp_gemm(sA, 160, P4, 5, b4, wv, lane);      // L4: 160->128
    frag_store_relu(sB, 128, 0, c, wv, lane, true); }
  __syncthreads();
  { v8f c = mlp_gemm(sB, 128, P5, 4, b5, wv, lane);      // L5: 128->128
    frag_store_relu(sA, 128, 0, c, wv, lane, true); }
  __syncthreads();
  v8f c0 = mlp_gemm(sA, 128, P6, 4, b6, wv, lane);       // L6: 128->256
  v8f c1 = mlp_gemm(sA, 128, P6, 4, b6, wv + 8, lane);
  // volume render: lane<16 holds pixel0 floors 0..7 across its 8 VGPRs
  int p = (lane < 16) ? 0 : 1;
  float a0 = 0.f, a1 = 0.f;
#pragma unroll
  for (int v = 0; v < 8; ++v) {
    float wt = s_w[p * 8 + v];
    a0 += wt * c0[v];
    a1 += wt * c1[v];
  }
  int pl = tile * 2 + p;
  int x = pl & 63, yy = (pl >> 6) & 63, b = pl >> 12;
  int n0 = wv * 16 + (lane & 15);
  size_t base = (((size_t)b * 256) * 64 + yy) * 64 + x;
  render[base + (size_t)n0 * 4096]         = 1.f / (1.f + __expf(-a0));
  render[base + (size_t)(n0 + 128) * 4096] = 1.f / (1.f + __expf(-a1));
}

// ---------------------------------------------------------------------------
// Direct conv3d 16->16 3x3x3 SAME on [8,16,8,64,64]
// ---------------------------------------------------------------------------
__global__ void conv3d_direct(const float* __restrict__ in, const float* __restrict__ w,
                              const float* __restrict__ bias, float* __restrict__ out) {
  __shared__ float lw[16 * 16 * 27];
  for (int i = threadIdx.x; i < 6912; i += blockDim.x) lw[i] = w[i];
  __syncthreads();
  int t  = blockIdx.x * blockDim.x + threadIdx.x;
  int x  = t & 63;
  int y  = (t >> 6) & 63;
  int f  = (t >> 12) & 7;
  int oc = (t >> 15) & 15;
  int b  = t >> 19;
  float acc = bias[oc];
  for (int ic = 0; ic < 16; ++ic) {
    const float* ip = in + (((size_t)b * 16 + ic) * 8) * 4096;
    const float* wq = lw + (oc * 16 + ic) * 27;
    for (int kf = 0; kf < 3; ++kf) {
      int fi = f + kf - 1; if ((unsigned)fi >= 8u) continue;
      for (int ky = 0; ky < 3; ++ky) {
        int yi = y + ky - 1; if ((unsigned)yi >= 64u) continue;
        for (int kx = 0; kx < 3; ++kx) {
          int xi = x + kx - 1; if ((unsigned)xi >= 64u) continue;
          acc += wq[kf * 9 + ky * 3 + kx] * ip[((size_t)fi * 64 + yi) * 64 + xi];
        }
      }
    }
  }
  out[t] = acc;
}

// ---------------------------------------------------------------------------
// Training-mode BN: per-channel mean/var, then normalize (+relu/+sigmoid)
// Layout assumed contiguous [outer][C][inner] (chStride==inner).
// ---------------------------------------------------------------------------
__global__ void bn_stats(const float* __restrict__ x, float* __restrict__ mv,
                         int chStride, int outer, int outerStride, int inner) {
  __shared__ float rs[256], rq[256];
  int c = blockIdx.x;
  float s = 0.f, q = 0.f;
  for (int o = 0; o < outer; ++o) {
    const float* p = x + (size_t)o * outerStride + (size_t)c * chStride;
    for (int i = threadIdx.x; i < inner; i += blockDim.x) {
      float v = p[i];
      s += v; q += v * v;
    }
  }
  rs[threadIdx.x] = s; rq[threadIdx.x] = q;
  __syncthreads();
  for (int st = 128; st > 0; st >>= 1) {
    if ((int)threadIdx.x < st) {
      rs[threadIdx.x] += rs[threadIdx.x + st];
      rq[threadIdx.x] += rq[threadIdx.x + st];
    }
    __syncthreads();
  }
  if (threadIdx.x == 0) {
    float n = (float)outer * (float)inner;
    float m = rs[0] / n;
    mv[2 * c]     = m;
    mv[2 * c + 1] = fmaxf(rq[0] / n - m * m, 0.f);
  }
}

__global__ void bn_apply(const float* __restrict__ x, float* __restrict__ y,
                         const float* __restrict__ mv, const float* __restrict__ g,
                         const float* __restrict__ bt, int C, int inner, long total,
                         int flags) {
  long idx = (long)blockIdx.x * blockDim.x + threadIdx.x;
  if (idx >= total) return;
  int c = (int)((idx / inner) % C);
  float m = mv[2 * c], v = mv[2 * c + 1];
  float val = (x[idx] - m) * rsqrtf(v + 1e-5f) * g[c] + bt[c];
  if (flags & 1) val = fmaxf(val, 0.f);
  if (flags & 2) val = 1.f / (1.f + __expf(-val));
  y[idx] = val;
}

__global__ void add_inplace(float* __restrict__ a, const float* __restrict__ b, long n) {
  long idx = (long)blockIdx.x * blockDim.x + threadIdx.x;
  if (idx < n) a[idx] += b[idx];
}

// ---------------------------------------------------------------------------
// Decoder tail conv: in [8,64,128,128] (2x nearest up) -> out [8,3,256,256]
// ---------------------------------------------------------------------------
__global__ void conv2d_up_small(const float* __restrict__ in, const float* __restrict__ w,
                                const float* __restrict__ bias, float* __restrict__ out) {
  __shared__ float lw[3 * 64 * 9];
  for (int i = threadIdx.x; i < 1728; i += blockDim.x) lw[i] = w[i];
  __syncthreads();
  long t = (long)blockIdx.x * blockDim.x + threadIdx.x;
  int x = (int)(t & 255);
  int y = (int)((t >> 8) & 255);
  long r = t >> 16;
  int oc = (int)(r % 3);
  int b  = (int)(r / 3);
  float acc = bias[oc];
  for (int ic = 0; ic < 64; ++ic) {
    const float* ip = in + ((size_t)b * 64 + ic) * 16384;
    const float* wq = lw + (oc * 64 + ic) * 9;
    for (int ky = 0; ky < 3; ++ky) {
      int yo = y + ky - 1; if ((unsigned)yo >= 256u) continue;
      int yi = yo >> 1;
      for (int kx = 0; kx < 3; ++kx) {
        int xo = x + kx - 1; if ((unsigned)xo >= 256u) continue;
        acc += wq[ky * 3 + kx] * ip[yi * 128 + (xo >> 1)];
      }
    }
  }
  out[t] = acc;
}

// ---------------------------------------------------------------------------
extern "C" void kernel_launch(void* const* d_in, const int* in_sizes, int n_in,
                              void* d_out, int out_size, void* d_ws, size_t ws_size,
                              hipStream_t stream) {
  (void)in_sizes; (void)n_in; (void)out_size; (void)ws_size;
  auto F32 = [&](int i) { return (const float*)d_in[i]; };

  const float* feature = F32(0);
  const float* rgbw = F32(1);  const float* rgbb = F32(2);
  const float* sgw  = F32(3);  const float* sgb  = F32(4);
  // res blocks: base 5 + 10*i : g1,b1,c1w,c1b,g2,b2,c2w,c2b,bng,bnb
  const float* l1w  = F32(35); const float* l1b  = F32(36);
  const float* l2w  = F32(37); const float* l2b  = F32(38);
  const float* l31w = F32(39); const float* l31b = F32(40);
  const float* l32w = F32(41); const float* l32b = F32(42);
  const float* l33w = F32(43); const float* l33b = F32(44);
  const float* l4w  = F32(45); const float* l4b  = F32(46);
  const float* l5w  = F32(47); const float* l5b  = F32(48);
  const float* l6w  = F32(49); const float* l6b  = F32(50);
  const float* u1w  = F32(51); const float* u1b  = F32(52);
  const float* u1g  = F32(53); const float* u1bt = F32(54);
  const float* u2w  = F32(55); const float* u2b  = F32(56);
  const float* u2g  = F32(57); const float* u2bt = F32(58);

  float* out1 = (float*)d_out;          // render   [8,256,64,64]
  float* out2 = out1 + 8388608;         // mini_pred[8,3,256,256]
  float* out3 = out2 + 1572864;         // sigma_map[8,16,8,64,64]

  // workspace arena
  unsigned short* P = (unsigned short*)d_ws;
  unsigned short* P_rgb = P; P += 294912;
  unsigned short* P_sig = P; P += 294912;
  unsigned short* P_l1  = P; P += 4096;
  unsigned short* P_l2  = P; P += 16384;
  unsigned short* P_l32 = P; P += 4096;
  unsigned short* P_l33 = P; P += 4096;
  unsigned short* P_l4  = P; P += 20480;
  unsigned short* P_l5  = P; P += 16384;
  unsigned short* P_l6  = P; P += 32768;
  unsigned short* P_u1  = P; P += 147456;
  float* Fb = (float*)(((uintptr_t)P + 255) & ~(uintptr_t)255);
  float* rf = Fb; Fb += 4194304;   // [8,128,64,64]
  float* hm = Fb; Fb += 4194304;   // [8,16,8,64,64]
  float* tA = Fb; Fb += 4194304;
  float* tB = Fb; Fb += 4194304;
  float* mv = Fb; Fb += 256;
  float* u1out = tA;               // alias tA+tB: [8,64,128,128]
  float* u2tmp = hm;               // alias: [8,3,256,256]

  auto packLin = [&](const float* W, unsigned short* dst, int K, int Kpad, int Nt) {
    int KBp = Kpad >> 5;
    int total = Nt * KBp * 512;
    pack_w<<<(total + 255) / 256, 256, 0, stream>>>(W, dst, K, KBp, total);
  };
  auto packConv = [&](const float* W, unsigned short* dst, int Nt) {
    int KBp = 72;  // 2304/32
    int total = Nt * KBp * 512;
    pack_conv<<<(total + 255) / 256, 256, 0, stream>>>(W, dst, KBp, total);
  };
  packConv(rgbw, P_rgb, 8);
  packConv(sgw,  P_sig, 8);
  packLin(l1w,  P_l1,  16,  32,  8);
  packLin(l2w,  P_l2,  128, 128, 8);
  packLin(l32w, P_l32, 128, 128, 2);
  packLin(l33w, P_l33, 16,  32,  8);
  packLin(l4w,  P_l4,  160, 160, 8);
  packLin(l5w,  P_l5,  128, 128, 8);
  packLin(l6w,  P_l6,  128, 128, 16);
  packConv(u1w, P_u1, 4);

  // feature -> rf / hm (conv2d 256->128, WMMA, M-tile 64)
  conv2d_wmma<256><<<512, 256, 0, stream>>>(feature, P_rgb, rgbb, rf,
                                            64, 64, 128, 64, 64, 0);
  conv2d_wmma<256><<<512, 256, 0, stream>>>(feature, P_sig, sgb, hm,
                                            64, 64, 128, 64, 64, 0);

  // 3 residual 3D blocks
  for (int i = 0; i < 3; ++i) {
    int base = 5 + 10 * i;
    const float* g1  = F32(base + 0); const float* bb1 = F32(base + 1);
    const float* c1w = F32(base + 2); const float* c1b = F32(base + 3);
    const float* g2  = F32(base + 4); const float* bb2 = F32(base + 5);
    const float* c2w = F32(base + 6); const float* c2b = F32(base + 7);
    const float* bng = F32(base + 8); const float* bnb = F32(base + 9);

    bn_stats<<<16, 256, 0, stream>>>(hm, mv, 32768, 8, 524288, 32768);
    bn_apply<<<16384, 256, 0, stream>>>(hm, tA, mv, g1, bb1, 16, 32768, 4194304L, 1);
    conv3d_direct<<<16384, 256, 0, stream>>>(tA, c1w, c1b, tB);
    bn_stats<<<16, 256, 0, stream>>>(tB, mv, 32768, 8, 524288, 32768);
    bn_apply<<<16384, 256, 0, stream>>>(tB, tA, mv, g2, bb2, 16, 32768, 4194304L, 1);
    conv3d_direct<<<16384, 256, 0, stream>>>(tA, c2w, c2b, tB);
    add_inplace<<<16384, 256, 0, stream>>>(hm, tB, 4194304L);
    bn_stats<<<16, 256, 0, stream>>>(hm, mv, 32768, 8, 524288, 32768);
    float* dst = (i == 2) ? out3 : hm;   // final BN -> sigma_map output
    bn_apply<<<16384, 256, 0, stream>>>(hm, dst, mv, bng, bnb, 16, 32768, 4194304L, 0);
  }

  // fused MLP + volume render -> render (out1)
  mlp_render<<<16384, 256, 0, stream>>>(out3, rf,
                                        P_l1, l1b, P_l2, l2b, l31w, l31b,
                                        P_l32, l32b, P_l33, l33b,
                                        P_l4, l4b, P_l5, l5b, P_l6, l6b, out1);

  // decoder UpBlock 1: 2x-up + conv 256->64 (WMMA, upsample folded) + BN + relu
  conv2d_wmma<128><<<2048, 128, 0, stream>>>(out1, P_u1, u1b, u1out,
                                             64, 64, 64, 128, 128, 1);
  bn_stats<<<64, 256, 0, stream>>>(u1out, mv, 16384, 8, 1048576, 16384);
  bn_apply<<<32768, 256, 0, stream>>>(u1out, u1out, mv, u1g, u1bt, 64, 16384,
                                      8388608L, 1);

  // decoder UpBlock 2: 2x-up + conv 64->3 + BN + relu + sigmoid -> mini_pred
  conv2d_up_small<<<6144, 256, 0, stream>>>(u1out, u2w, u2b, u2tmp);
  bn_stats<<<3, 256, 0, stream>>>(u2tmp, mv, 65536, 8, 196608, 65536);
  bn_apply<<<6144, 256, 0, stream>>>(u2tmp, out2, mv, u2g, u2bt, 3, 65536,
                                     1572864L, 3);
}
